// CausalSelfAttentionTF_40862318854314
// MI455X (gfx1250) — matrix operand order repaired and checked by reading, not compiled, not run on hardware
//
#include <hip/hip_runtime.h>

typedef __attribute__((ext_vector_type(16))) _Float16 v16h;
typedef __attribute__((ext_vector_type(8)))  _Float16 v8h;
typedef __attribute__((ext_vector_type(8)))  float    v8f;

namespace {
constexpr int kDModel  = 768;
constexpr int kNHead   = 12;
constexpr int kHeadDim = 64;
constexpr int kSeq     = 1024;
constexpr int kBatch   = 8;
constexpr int kRows    = kBatch * kSeq;   // 8192
constexpr int kQkvN    = 3 * kDModel;     // 2304
}

// ---------------- WMMA helpers ----------------

__device__ __forceinline__ v8f wmma16(v16h a, v16h b, v8f c) {
  // v_wmma_f32_16x16x32_f16  D = A(16x32) * B(32x16) + C(16x16 f32)
  return __builtin_amdgcn_wmma_f32_16x16x32_f16(false, a, false, b, (short)0, c,
                                                false, false);
}

__device__ __forceinline__ v8f zero8() {
  v8f z;
#pragma unroll
  for (int i = 0; i < 8; ++i) z[i] = 0.0f;
  return z;
}

// A fragment (16 rows x 32 K, f16), row-major source with leading dim ld.
// Layout: lanes 0-15 row=lane, K=0..15 consecutive;
//         lanes 16-31 same rows, K=16..31.
__device__ __forceinline__ v16h load_fragA(const _Float16* __restrict__ base, int ld) {
  const int lane = (int)(threadIdx.x & 31u);
  const int row  = lane & 15;
  const int kb   = (lane & 16) ? 16 : 0;
  const _Float16* p = base + row * ld + kb;
  v8h lo = *reinterpret_cast<const v8h*>(p);
  v8h hi = *reinterpret_cast<const v8h*>(p + 8);
  return __builtin_shufflevector(lo, hi, 0,1,2,3,4,5,6,7,8,9,10,11,12,13,14,15);
}

// B fragment (32 K x 16 cols, f16) loaded from B^T stored row-major [N][K].
// Same contiguous k order as load_fragA: lanes 0-15 col=lane, K=0..15 consecutive; lanes 16-31 K=16..31.
__device__ __forceinline__ v16h load_fragB(const _Float16* __restrict__ base, int ld) {
  const int lane = (int)(threadIdx.x & 31u);
  const int col  = lane & 15;
  const int kb   = (lane & 16) ? 16 : 0;
  const _Float16* p = base + col * ld + kb;
  v8h lo = *reinterpret_cast<const v8h*>(p);
  v8h hi = *reinterpret_cast<const v8h*>(p + 8);
  return __builtin_shufflevector(lo, hi, 0,1,2,3,4,5,6,7,8,9,10,11,12,13,14,15);
}

// ---------------- conversion kernels ----------------

__global__ void k_cvt_f32_to_f16(const float* __restrict__ src,
                                 _Float16* __restrict__ dst, int n) {
  int i = blockIdx.x * blockDim.x + threadIdx.x;
  if (i < n) dst[i] = (_Float16)src[i];
}

// W is [K=768][N] row-major fp32; Wt is [N][768] f16 (i.e. W^T).
__global__ void k_cvt_wT(const float* __restrict__ W, _Float16* __restrict__ Wt,
                         int N, int total) {
  int i = blockIdx.x * blockDim.x + threadIdx.x;
  if (i >= total) return;
  int nIdx = i / kDModel;   // output column (row of Wt)
  int cIdx = i - nIdx * kDModel;
  Wt[i] = (_Float16)W[cIdx * N + nIdx];
}

// ---------------- QKV projection GEMM ----------------
// xh [8192][768] f16  x  W_attn [768][2304]  -> q (scaled)/k f16 [B,H,T,64], vT f16 [B,H,64,T]

__device__ __forceinline__ void qkv_store(v8f acc, int col, int rowbase, int half,
                                          const float* __restrict__ b_attn,
                                          _Float16* __restrict__ q,
                                          _Float16* __restrict__ k,
                                          _Float16* __restrict__ vT) {
  const float bias = b_attn[col];
#pragma unroll
  for (int e = 0; e < 8; ++e) {
    float val = acc[e] + bias;
    int gr = rowbase + e + half * 8;         // global row = b*1024 + t
    int bb = gr >> 10;
    int tt = gr & 1023;
    if (col < kDModel) {                     // Q (fold softmax scale 1/sqrt(64))
      int h = col >> 6, d = col & 63;
      q[(((bb * kNHead + h) << 10) + tt) * kHeadDim + d] = (_Float16)(val * 0.125f);
    } else if (col < 2 * kDModel) {          // K
      int c2 = col - kDModel;
      int h = c2 >> 6, d = c2 & 63;
      k[(((bb * kNHead + h) << 10) + tt) * kHeadDim + d] = (_Float16)val;
    } else {                                 // V, stored transposed [B,H,hd,T]
      int c2 = col - 2 * kDModel;
      int h = c2 >> 6, d = c2 & 63;
      vT[((bb * kNHead + h) * kHeadDim + d) * kSeq + tt] = (_Float16)val;
    }
  }
}

__global__ void k_qkv(const _Float16* __restrict__ xh,
                      const _Float16* __restrict__ Wah,  // [2304][768] = W_attn^T f16
                      const float* __restrict__ b_attn,
                      _Float16* __restrict__ q, _Float16* __restrict__ k,
                      _Float16* __restrict__ vT) {
  const int wave = blockIdx.x * (blockDim.x >> 5) + ((int)threadIdx.x >> 5);
  const int NT = kQkvN / 64;                 // 36
  const int nt = wave % NT;
  const int mt = wave / NT;                  // 0..511
  const int rowbase = mt * 16;
  const int colbase = nt * 64;
  const int lane = (int)(threadIdx.x & 31u);
  const int m = lane & 15;
  const int half = (lane >> 4) & 1;

  const _Float16* Arow = xh + (size_t)rowbase * kDModel;
  const _Float16* B0 = Wah + (size_t)(colbase +  0) * kDModel;
  const _Float16* B1 = Wah + (size_t)(colbase + 16) * kDModel;
  const _Float16* B2 = Wah + (size_t)(colbase + 32) * kDModel;
  const _Float16* B3 = Wah + (size_t)(colbase + 48) * kDModel;

  // software pipeline: fragments for chunk kc are loaded during chunk kc-32's WMMAs
  v16h af = load_fragA(Arow, kDModel);
  v16h c0 = load_fragB(B0, kDModel);
  v16h c1 = load_fragB(B1, kDModel);
  v16h c2 = load_fragB(B2, kDModel);
  v16h c3 = load_fragB(B3, kDModel);

  v8f a0 = zero8(), a1 = zero8(), a2 = zero8(), a3 = zero8();
  for (int kc = 32; kc < kDModel; kc += 32) {
    v16h afn = load_fragA(Arow + kc, kDModel);
    v16h n0  = load_fragB(B0 + kc, kDModel);
    v16h n1  = load_fragB(B1 + kc, kDModel);
    v16h n2  = load_fragB(B2 + kc, kDModel);
    v16h n3  = load_fragB(B3 + kc, kDModel);
    a0 = wmma16(af, c0, a0);
    a1 = wmma16(af, c1, a1);
    a2 = wmma16(af, c2, a2);
    a3 = wmma16(af, c3, a3);
    af = afn; c0 = n0; c1 = n1; c2 = n2; c3 = n3;
  }
  a0 = wmma16(af, c0, a0);
  a1 = wmma16(af, c1, a1);
  a2 = wmma16(af, c2, a2);
  a3 = wmma16(af, c3, a3);

  qkv_store(a0, colbase +  0 + m, rowbase, half, b_attn, q, k, vT);
  qkv_store(a1, colbase + 16 + m, rowbase, half, b_attn, q, k, vT);
  qkv_store(a2, colbase + 32 + m, rowbase, half, b_attn, q, k, vT);
  qkv_store(a3, colbase + 48 + m, rowbase, half, b_attn, q, k, vT);
}

// ---------------- flash attention (one wave per 16-query tile) ----------------
// S^T = K * Q^T  (lane axis = query, VGPR axis = key)
// O^T = V^T * P^T accumulated with online-softmax rescaling; y merged heads f16.

__global__ void k_attn(const _Float16* __restrict__ q,
                       const _Float16* __restrict__ k,
                       const _Float16* __restrict__ vT,
                       _Float16* __restrict__ y) {
  const int wave = blockIdx.x * (blockDim.x >> 5) + ((int)threadIdx.x >> 5);
  const int qt = wave & 63;                  // T/16 = 64 query tiles
  const int bh = wave >> 6;                  // 0..95 (batch*head)
  const int qbase = qt * 16;
  const int lane = (int)(threadIdx.x & 31u);
  const int m = lane & 15;                   // this lane's query row within tile
  const int half = (lane >> 4) & 1;
  const int qglob = qbase + m;

  // Q^T as B operand: lane m holds q[m][half*16 ..], two K=32 chunks over hd=64
  const _Float16* qp = q + (size_t)(bh * kSeq + qbase) * kHeadDim;
  const v16h qB0 = load_fragB(qp, kHeadDim);        // d 0..31
  const v16h qB1 = load_fragB(qp + 32, kHeadDim);   // d 32..63

  v8f o0 = zero8(), o1 = zero8(), o2 = zero8(), o3 = zero8();
  float mrow = -1e30f, lrow = 0.0f;

  for (int j = 0; j <= qbase + 15; j += 32) {
    // ---- issue ALL loads for this chunk up front (K tiles + V tiles);
    //      V loads are independent of softmax and cover its latency.
    const _Float16* kp0 = k + (size_t)(bh * kSeq + j) * kHeadDim;
    const _Float16* kp1 = kp0 + 16 * kHeadDim;
    const _Float16* vp  = vT + (size_t)(bh * kHeadDim) * kSeq + j;
    v16h ka00 = load_fragA(kp0, kHeadDim);
    v16h ka01 = load_fragA(kp0 + 32, kHeadDim);
    v16h ka10 = load_fragA(kp1, kHeadDim);
    v16h ka11 = load_fragA(kp1 + 32, kHeadDim);
    v16h va0 = load_fragA(vp + 0 * 16 * kSeq, kSeq);
    v16h va1 = load_fragA(vp + 1 * 16 * kSeq, kSeq);
    v16h va2 = load_fragA(vp + 2 * 16 * kSeq, kSeq);
    v16h va3 = load_fragA(vp + 3 * 16 * kSeq, kSeq);

    // ---- S^T tiles: keys [j, j+16) and [j+16, j+32)
    v8f s0 = zero8(), s1 = zero8();
    s0 = wmma16(ka00, qB0, s0);
    s0 = wmma16(ka01, qB1, s0);
    s1 = wmma16(ka10, qB0, s1);
    s1 = wmma16(ka11, qB1, s1);

    // ---- causal mask (key index on VGPR axis: e + half*8 within tile)
#pragma unroll
    for (int e = 0; e < 8; ++e) {
      int key0 = j + e + half * 8;
      if (key0 > qglob)      s0[e] = -1e9f;
      if (key0 + 16 > qglob) s1[e] = -1e9f;
    }
    // ---- online softmax stats (reduce in-lane + across half-waves)
    float tmax = -1e30f;
#pragma unroll
    for (int e = 0; e < 8; ++e) tmax = fmaxf(tmax, fmaxf(s0[e], s1[e]));
    tmax = fmaxf(tmax, __shfl_xor(tmax, 16, 32));
    const float mnew  = fmaxf(mrow, tmax);
    const float alpha = __expf(mrow - mnew);

    float p0[8], p1[8];
    float lsum = 0.0f;
#pragma unroll
    for (int e = 0; e < 8; ++e) {
      p0[e] = __expf(s0[e] - mnew);
      p1[e] = __expf(s1[e] - mnew);
      lsum += p0[e] + p1[e];
    }
    lsum += __shfl_xor(lsum, 16, 32);
    lrow = lrow * alpha + lsum;
    mrow = mnew;

    // ---- build P^T B-fragment: lane m holds keys j+half*16 .. +15 for column m
    float x0[8], x1[8];
#pragma unroll
    for (int e = 0; e < 8; ++e) x0[e] = __shfl_xor(p0[e], 16, 32);
#pragma unroll
    for (int e = 0; e < 8; ++e) x1[e] = __shfl_xor(p1[e], 16, 32);
    v16h pB;
#pragma unroll
    for (int e = 0; e < 8; ++e) {
      pB[e]     = (_Float16)(half ? x1[e] : p0[e]);   // keys +0..7 of this half's 16
      pB[e + 8] = (_Float16)(half ? p1[e] : x0[e]);   // keys +8..15
    }
    // ---- rescale O and accumulate O^T += V^T * P^T  (4 d-tiles of 16)
#pragma unroll
    for (int e = 0; e < 8; ++e) {
      o0[e] *= alpha; o1[e] *= alpha; o2[e] *= alpha; o3[e] *= alpha;
    }
    o0 = wmma16(va0, pB, o0);
    o1 = wmma16(va1, pB, o1);
    o2 = wmma16(va2, pB, o2);
    o3 = wmma16(va3, pB, o3);
  }

  // ---- normalize and store merged-head y (f16). O^T: lane=query col, VGPRs=d.
  const float inv = 1.0f / lrow;
  const int b = bh / kNHead;
  const int h = bh - b * kNHead;
  _Float16* yp = y + (size_t)(b * kSeq + qglob) * kDModel + h * kHeadDim + half * 8;
  v8h r0, r1, r2, r3;
#pragma unroll
  for (int e = 0; e < 8; ++e) {
    r0[e] = (_Float16)(o0[e] * inv);
    r1[e] = (_Float16)(o1[e] * inv);
    r2[e] = (_Float16)(o2[e] * inv);
    r3[e] = (_Float16)(o3[e] * inv);
  }
  *reinterpret_cast<v8h*>(yp +  0) = r0;
  *reinterpret_cast<v8h*>(yp + 16) = r1;
  *reinterpret_cast<v8h*>(yp + 32) = r2;
  *reinterpret_cast<v8h*>(yp + 48) = r3;
}

// ---------------- output projection GEMM ----------------
// y [8192][768] f16 x W_proj [768][768] -> out fp32 + b_proj

__global__ void k_proj(const _Float16* __restrict__ y,
                       const _Float16* __restrict__ Wph,  // [768][768] = W_proj^T f16
                       const float* __restrict__ b_proj,
                       float* __restrict__ out) {
  const int wave = blockIdx.x * (blockDim.x >> 5) + ((int)threadIdx.x >> 5);
  const int NT = kDModel / 64;               // 12
  const int nt = wave % NT;
  const int mt = wave / NT;                  // 0..511
  const int rowbase = mt * 16;
  const int colbase = nt * 64;
  const int lane = (int)(threadIdx.x & 31u);
  const int m = lane & 15;
  const int half = (lane >> 4) & 1;

  const _Float16* Arow = y + (size_t)rowbase * kDModel;
  const _Float16* B0 = Wph + (size_t)(colbase +  0) * kDModel;
  const _Float16* B1 = Wph + (size_t)(colbase + 16) * kDModel;
  const _Float16* B2 = Wph + (size_t)(colbase + 32) * kDModel;
  const _Float16* B3 = Wph + (size_t)(colbase + 48) * kDModel;

  v16h af = load_fragA(Arow, kDModel);
  v16h c0 = load_fragB(B0, kDModel);
  v16h c1 = load_fragB(B1, kDModel);
  v16h c2 = load_fragB(B2, kDModel);
  v16h c3 = load_fragB(B3, kDModel);

  v8f a0 = zero8(), a1 = zero8(), a2 = zero8(), a3 = zero8();
  for (int kc = 32; kc < kDModel; kc += 32) {
    v16h afn = load_fragA(Arow + kc, kDModel);
    v16h n0  = load_fragB(B0 + kc, kDModel);
    v16h n1  = load_fragB(B1 + kc, kDModel);
    v16h n2  = load_fragB(B2 + kc, kDModel);
    v16h n3  = load_fragB(B3 + kc, kDModel);
    a0 = wmma16(af, c0, a0);
    a1 = wmma16(af, c1, a1);
    a2 = wmma16(af, c2, a2);
    a3 = wmma16(af, c3, a3);
    af = afn; c0 = n0; c1 = n1; c2 = n2; c3 = n3;
  }
  a0 = wmma16(af, c0, a0);
  a1 = wmma16(af, c1, a1);
  a2 = wmma16(af, c2, a2);
  a3 = wmma16(af, c3, a3);

  v8f acc[4] = {a0, a1, a2, a3};
#pragma unroll
  for (int c = 0; c < 4; ++c) {
    const int col = colbase + c * 16 + m;
    const float bias = b_proj[col];
#pragma unroll
    for (int e = 0; e < 8; ++e) {
      int gr = rowbase + e + half * 8;
      out[(size_t)gr * kDModel + col] = acc[c][e] + bias;
    }
  }
}

// ---------------- launcher ----------------

extern "C" void kernel_launch(void* const* d_in, const int* in_sizes, int n_in,
                              void* d_out, int out_size, void* d_ws, size_t ws_size,
                              hipStream_t stream) {
  (void)in_sizes; (void)n_in; (void)out_size; (void)ws_size;
  const float* x      = (const float*)d_in[0];
  const float* W_attn = (const float*)d_in[1];
  const float* b_attn = (const float*)d_in[2];
  const float* W_proj = (const float*)d_in[3];
  const float* b_proj = (const float*)d_in[4];
  float* out = (float*)d_out;

  _Float16* ws = (_Float16*)d_ws;
  size_t off = 0;
  _Float16* xh  = ws + off; off += (size_t)kRows * kDModel;
  _Float16* qb  = ws + off; off += (size_t)kBatch * kNHead * kSeq * kHeadDim;
  _Float16* kb  = ws + off; off += (size_t)kBatch * kNHead * kSeq * kHeadDim;
  _Float16* vT  = ws + off; off += (size_t)kBatch * kNHead * kSeq * kHeadDim;
  _Float16* yb  = ws + off; off += (size_t)kRows * kDModel;
  _Float16* Wah = ws + off; off += (size_t)kQkvN * kDModel;
  _Float16* Wph = ws + off; off += (size_t)kDModel * kDModel;
  // total ~67.6 MB of workspace

  int n;
  n = kRows * kDModel;
  k_cvt_f32_to_f16<<<(n + 255) / 256, 256, 0, stream>>>(x, xh, n);
  n = kDModel * kQkvN;
  k_cvt_wT<<<(n + 255) / 256, 256, 0, stream>>>(W_attn, Wah, kQkvN, n);
  n = kDModel * kDModel;
  k_cvt_wT<<<(n + 255) / 256, 256, 0, stream>>>(W_proj, Wph, kDModel, n);

  // QKV GEMM: 512 row-tiles x 36 col-tiles = 18432 waves / 8 per block
  k_qkv<<<2304, 256, 0, stream>>>(xh, Wah, b_attn, qb, kb, vT);
  // Attention: 96 (b,h) x 64 q-tiles = 6144 waves
  k_attn<<<768, 256, 0, stream>>>(qb, kb, vT, yb);
  // Output projection: 512 x 12 = 6144 waves
  k_proj<<<768, 256, 0, stream>>>(yb, Wph, b_proj, out);
}
